// PatchAttention_4595615006864
// MI455X (gfx1250) — matrix operand order repaired and checked by reading, not compiled, not run on hardware
//
#include <hip/hip_runtime.h>

#define DEV __device__ __forceinline__

constexpr int NB  = 2;    // batch
constexpr int NN  = 384;  // tokens
constexpr int ND  = 256;  // model dim
constexpr int NH  = 8;    // heads
constexpr int NHD = 32;   // head dim
constexpr int NQKV = 3 * ND;  // 768
constexpr float QSCALE = 0.17677669529663687f;  // 1/sqrt(32)

typedef __bf16 v16bf __attribute__((ext_vector_type(16)));
typedef float  v8f   __attribute__((ext_vector_type(8)));

DEV v8f wmma_bf16(v16bf a, v16bf b, v8f c) {
  // emits v_wmma_f32_16x16x32_bf16
  return __builtin_amdgcn_wmma_f32_16x16x32_bf16(false, a, false, b, (short)0, c,
                                                 false, false);
}

// ---------- wave32 WMMA fragment loaders (16x16x32 bf16) ----------
// A[m][k] 16x32, row-major source with 'stride' elements between rows.
// Layout: lane(m=lane&15); lanes<16 hold K=0..15, lanes>=16 hold K=16..31.
DEV v16bf load_a(const __bf16* src, int stride) {
  int lane = threadIdx.x & 31;
  int m = lane & 15;
  int kb = (lane < 16) ? 0 : 16;
  v16bf a;
#pragma unroll
  for (int e = 0; e < 16; ++e) a[e] = src[m * stride + kb + e];
  return a;
}

DEV v16bf load_a_pad(const __bf16* src, int stride, int valid_m) {
  int lane = threadIdx.x & 31;
  int m = lane & 15;
  int kb = (lane < 16) ? 0 : 16;
  v16bf a;
  if (m < valid_m) {
#pragma unroll
    for (int e = 0; e < 16; ++e) a[e] = src[m * stride + kb + e];
  } else {
#pragma unroll
    for (int e = 0; e < 16; ++e) a[e] = (__bf16)0.f;
  }
  return a;
}

// B[k][n] 32x16 from row-major KxN source ('stride' elems between k rows).
// Layout (contiguous k, not ISA table order): lane(n=lane&15); lanes<16 hold K=0..15, lanes>=16 hold K=16..31.
DEV v16bf load_b_kn(const __bf16* src, int stride) {
  int lane = threadIdx.x & 31;
  int n = lane & 15;
  int kb = (lane < 16) ? 0 : 16;
  v16bf b;
#pragma unroll
  for (int e = 0; e < 16; ++e) b[e] = src[(kb + e) * stride + n];
  return b;
}

// B = W^T from row-major NxK weight W[n][k].
DEV v16bf load_b_nk(const __bf16* src, int stride) {
  int lane = threadIdx.x & 31;
  int n = lane & 15;
  int kb = (lane < 16) ? 0 : 16;
  v16bf b;
#pragma unroll
  for (int e = 0; e < 16; ++e) b[e] = src[n * stride + kb + e];
  return b;
}

DEV v16bf load_b_nk_pad(const __bf16* src, int stride, int valid_n) {
  int lane = threadIdx.x & 31;
  int n = lane & 15;
  int kb = (lane < 16) ? 0 : 16;
  v16bf b;
  if (n < valid_n) {
#pragma unroll
    for (int e = 0; e < 16; ++e) b[e] = src[n * stride + kb + e];
  } else {
#pragma unroll
    for (int e = 0; e < 16; ++e) b[e] = (__bf16)0.f;
  }
  return b;
}

DEV float silu(float x) { return x / (1.0f + __expf(-x)); }

// ---------- K0: bf16 conversions + rv_w2 transpose ----------
__global__ __launch_bounds__(256) void pa_convert(
    const float* __restrict__ x, const float* __restrict__ qkv_w,
    const float* __restrict__ rb_w2, const float* __restrict__ rv_w2,
    __bf16* __restrict__ xb, __bf16* __restrict__ wqkvb,
    __bf16* __restrict__ rbw2b, float* __restrict__ w2t) {
  int i = blockIdx.x * blockDim.x + threadIdx.x;
  if (i < NB * NN * ND) xb[i] = (__bf16)x[i];
  if (i < NQKV * ND) wqkvb[i] = (__bf16)qkv_w[i];
  if (i < NH * ND) rbw2b[i] = (__bf16)rb_w2[i];
  if (i < ND * ND) {
    int o = i / ND, c = i % ND;
    w2t[c * ND + o] = rv_w2[i];  // transpose for coalesced matvec
  }
}

// ---------- K1: QKV projection (WMMA, K=256), head-major bf16 out ----------
__global__ __launch_bounds__(32) void pa_qkv(
    const __bf16* __restrict__ xb, const __bf16* __restrict__ wqkvb,
    const float* __restrict__ qkv_b, __bf16* __restrict__ Qb,
    __bf16* __restrict__ Kb, __bf16* __restrict__ Vb) {
  int bid = blockIdx.x;
  int ot = bid % 48; bid /= 48;           // 48 output tiles of 16 over 768
  int it = bid % 24; int b = bid / 24;    // 24 row tiles of 16 over 384
  int lane = threadIdx.x & 31;
  int i0 = it * 16, o0 = ot * 16;
  v8f acc = {};
#pragma unroll
  for (int kt = 0; kt < 8; ++kt) {
    v16bf a  = load_a(xb + (b * NN + i0) * ND + kt * 32, ND);
    v16bf bm = load_b_nk(wqkvb + o0 * ND + kt * 32, ND);
    acc = wmma_bf16(a, bm, acc);
  }
  int n = lane & 15;
  int o = o0 + n;
  int which = o >> 8;             // 0=q 1=k 2=v (uniform per tile)
  int o8 = o & 255;
  int h = o8 >> 5, dp = o8 & 31;
  __bf16* dst = (which == 0) ? Qb : ((which == 1) ? Kb : Vb);
  float scale = (which == 0) ? QSCALE : 1.0f;
  float bias = qkv_b[o];
#pragma unroll
  for (int r = 0; r < 8; ++r) {
    int m = (lane < 16) ? r : 8 + r;
    dst[((b * NH + h) * NN + (i0 + m)) * NHD + dp] = (__bf16)((acc[r] + bias) * scale);
  }
}

// ---------- K2: scores = QK^T (prescaled) + rel_bias MLP ----------
// one block per (b, i-tile16, j-tile16); 8 waves = 8 heads
__global__ __launch_bounds__(256) void pa_scores(
    const float* __restrict__ coords, const float* __restrict__ rb_w1,
    const float* __restrict__ rb_b1, const float* __restrict__ rb_b2,
    const __bf16* __restrict__ rbw2b, const __bf16* __restrict__ Qb,
    const __bf16* __restrict__ Kb, float* __restrict__ Sg) {
  __shared__ __bf16 hid[256 * 32];     // 256 pairs x 32-channel chunk (16 KB)
  __shared__ float biasBuf[256 * 8];   // per-pair, per-head bias (8 KB)
  int bid = blockIdx.x;
  int jt = bid % 24; bid /= 24;
  int it = bid % 24; int b = bid / 24;
  int i0 = it * 16, j0 = jt * 16;
  int tid = threadIdx.x, lane = tid & 31, w = tid >> 5;  // w == head

  // QK^T for head w: one WMMA (K = HD = 32)
  v16bf qa = load_a(Qb + ((b * NH + w) * NN + i0) * NHD, NHD);
  v16bf kf = load_b_nk(Kb + ((b * NH + w) * NN + j0) * NHD, NHD);
  v8f sqk = {};
  sqk = wmma_bf16(qa, kf, sqk);

  // rel_bias: layer1 (3->256) on VALU in 32-channel chunks, layer2 via WMMA
  v8f accB0 = {}, accB1 = {};
  int cl = tid & 31;     // channel-in-chunk
  int pb = tid >> 5;     // pair base
  for (int kt = 0; kt < 8; ++kt) {
    __syncthreads();     // protect previous chunk's WMMA reads
    int c = kt * 32 + cl;
    float w0 = rb_w1[c * 3 + 0], w1 = rb_w1[c * 3 + 1], w2 = rb_w1[c * 3 + 2];
    float bb = rb_b1[c];
#pragma unroll 4
    for (int m = 0; m < 32; ++m) {
      int p = pb + m * 8;
      int il = p >> 4, jl = p & 15;
      const float* ci = coords + (b * NN + i0 + il) * 3;
      const float* cj = coords + (b * NN + j0 + jl) * 3;
      float hv = w0 * (ci[0] - cj[0]) + w1 * (ci[1] - cj[1]) + w2 * (ci[2] - cj[2]) + bb;
      hid[p * 32 + cl] = (__bf16)silu(hv);
    }
    __syncthreads();
    v16bf bw = load_b_nk_pad(rbw2b + kt * 32, ND, NH);  // 8 heads, padded to 16
    v16bf a0 = load_a(hid + (w * 2 + 0) * 16 * 32, 32);
    v16bf a1 = load_a(hid + (w * 2 + 1) * 16 * 32, 32);
    accB0 = wmma_bf16(a0, bw, accB0);
    accB1 = wmma_bf16(a1, bw, accB1);
  }
  // scatter bias to LDS: row = pair, col = head
  int hc = lane & 15;
#pragma unroll
  for (int r = 0; r < 8; ++r) {
    int pr = (lane < 16) ? r : 8 + r;
    if (hc < NH) {
      biasBuf[((w * 2 + 0) * 16 + pr) * 8 + hc] = accB0[r];
      biasBuf[((w * 2 + 1) * 16 + pr) * 8 + hc] = accB1[r];
    }
  }
  __syncthreads();
  float b2 = rb_b2[w];
#pragma unroll
  for (int r = 0; r < 8; ++r) {
    int il = (lane < 16) ? r : 8 + r;
    int jl = lane & 15;
    int p = il * 16 + jl;
    Sg[((size_t)(b * NH + w) * NN + (i0 + il)) * NN + (j0 + jl)] =
        sqk[r] + biasBuf[p * 8 + w] + b2;
  }
}

// ---------- K3: row softmax, fp32 scores -> bf16 weights ----------
__global__ __launch_bounds__(256) void pa_softmax(const float* __restrict__ Sg,
                                                  __bf16* __restrict__ Wgt) {
  int w = threadIdx.x >> 5, lane = threadIdx.x & 31;
  int row = blockIdx.x * 8 + w;  // (b*H+h)*N + i
  const float* s = Sg + (size_t)row * NN;
  float vals[12];
  float mx = -3.0e38f;
#pragma unroll
  for (int t = 0; t < 12; ++t) { vals[t] = s[lane + t * 32]; mx = fmaxf(mx, vals[t]); }
#pragma unroll
  for (int m = 16; m >= 1; m >>= 1) mx = fmaxf(mx, __shfl_xor(mx, m, 32));
  float sum = 0.f;
#pragma unroll
  for (int t = 0; t < 12; ++t) { vals[t] = __expf(vals[t] - mx); sum += vals[t]; }
#pragma unroll
  for (int m = 16; m >= 1; m >>= 1) sum += __shfl_xor(sum, m, 32);
  float inv = 1.0f / sum;
  __bf16* wr = Wgt + (size_t)row * NN;
#pragma unroll
  for (int t = 0; t < 12; ++t) wr[lane + t * 32] = (__bf16)(vals[t] * inv);
}

// ---------- K4: base = W @ V per (b,h,i-tile16); K=384, N=32 ----------
__global__ __launch_bounds__(64) void pa_base(const __bf16* __restrict__ Wgt,
                                              const __bf16* __restrict__ Vb,
                                              float* __restrict__ baseO) {
  int bid = blockIdx.x;
  int it = bid % 24; bid /= 24;
  int h = bid % NH; int b = bid / NH;
  int i0 = it * 16;
  int w = threadIdx.x >> 5, lane = threadIdx.x & 31;  // w = n-tile (0/1)
  const __bf16* Wr = Wgt + ((size_t)(b * NH + h) * NN + i0) * NN;
  const __bf16* Vh = Vb + (size_t)((b * NH + h) * NN) * NHD;
  v8f acc = {};
#pragma unroll
  for (int kt = 0; kt < 12; ++kt) {
    v16bf a  = load_a(Wr + kt * 32, NN);
    v16bf bm = load_b_kn(Vh + (kt * 32) * NHD + w * 16, NHD);
    acc = wmma_bf16(a, bm, acc);
  }
  int n = lane & 15;
#pragma unroll
  for (int r = 0; r < 8; ++r) {
    int m = (lane < 16) ? r : 8 + r;
    baseO[((size_t)(b * NH + h) * NN + (i0 + m)) * NHD + w * 16 + n] = acc[r];
  }
}

// ---------- K5: weighted hidden aggregation (softmax sums to 1 =>
//  rel_context = (sum_j w*silu(h1)) @ W2^T + b2) + base -> output ----------
// one block per (b,i); GEMM (8 heads padded to 16) x 384 j x 256 channels
__global__ __launch_bounds__(256) void pa_relctx(
    const float* __restrict__ coords, const float* __restrict__ rv_w1,
    const float* __restrict__ rv_b1, const float* __restrict__ rv_b2,
    const float* __restrict__ w2t, const __bf16* __restrict__ Wgt,
    const float* __restrict__ baseO, float* __restrict__ out) {
  __shared__ __bf16 hidv[32 * 256];   // 32-j chunk x 256 channels (16 KB)
  __shared__ float Hagg[8 * 256];     // aggregated hidden per head (8 KB)
  int b = blockIdx.x / NN, i = blockIdx.x % NN;
  int tid = threadIdx.x, lane = tid & 31, w = tid >> 5;
  float ci0 = coords[(b * NN + i) * 3 + 0];
  float ci1 = coords[(b * NN + i) * 3 + 1];
  float ci2 = coords[(b * NN + i) * 3 + 2];
  int c = tid;  // this thread's channel
  float w0 = rv_w1[c * 3 + 0], w1 = rv_w1[c * 3 + 1], w2 = rv_w1[c * 3 + 2];
  float bb = rv_b1[c];
  const __bf16* Ab = Wgt + (size_t)b * NH * NN * NN + (size_t)i * NN;  // A[m=h][k=j], stride NN*NN
  v8f acc0 = {}, acc1 = {};
  for (int kt = 0; kt < 12; ++kt) {
    __syncthreads();
#pragma unroll 4
    for (int jj = 0; jj < 32; ++jj) {
      int j = kt * 32 + jj;
      const float* cj = coords + (b * NN + j) * 3;
      float hv = w0 * (ci0 - cj[0]) + w1 * (ci1 - cj[1]) + w2 * (ci2 - cj[2]) + bb;
      hidv[jj * 256 + c] = (__bf16)silu(hv);
    }
    __syncthreads();
    v16bf a  = load_a_pad(Ab + kt * 32, NN * NN, NH);       // heads padded 8->16
    v16bf b0 = load_b_kn(hidv + (w * 2 + 0) * 16, 256);
    v16bf b1 = load_b_kn(hidv + (w * 2 + 1) * 16, 256);
    acc0 = wmma_bf16(a, b0, acc0);
    acc1 = wmma_bf16(a, b1, acc1);
  }
  int n = lane & 15;
#pragma unroll
  for (int r = 0; r < 8; ++r) {
    int m = (lane < 16) ? r : 8 + r;
    if (m < NH) {
      Hagg[m * 256 + (w * 2 + 0) * 16 + n] = acc0[r];
      Hagg[m * 256 + (w * 2 + 1) * 16 + n] = acc1[r];
    }
  }
  __syncthreads();
  // final per-head 256->32 matvec (W2 slice) + bias + base
  int o = tid;          // output feature 0..255
  int h = o >> 5;
  float accf = rv_b2[o];
  for (int cc = 0; cc < 256; ++cc) accf += Hagg[h * 256 + cc] * w2t[cc * 256 + o];
  out[((size_t)b * NN + i) * ND + o] =
      baseO[((size_t)(b * NH + h) * NN + i) * NHD + (o & 31)] + accf;
}

extern "C" void kernel_launch(void* const* d_in, const int* in_sizes, int n_in,
                              void* d_out, int out_size, void* d_ws, size_t ws_size,
                              hipStream_t stream) {
  const float* x      = (const float*)d_in[0];
  const float* coords = (const float*)d_in[1];
  const float* qkv_w  = (const float*)d_in[2];
  const float* qkv_b  = (const float*)d_in[3];
  const float* rb_w1  = (const float*)d_in[4];
  const float* rb_b1  = (const float*)d_in[5];
  const float* rb_w2  = (const float*)d_in[6];
  const float* rb_b2  = (const float*)d_in[7];
  const float* rv_w1  = (const float*)d_in[8];
  const float* rv_b1  = (const float*)d_in[9];
  const float* rv_w2  = (const float*)d_in[10];
  const float* rv_b2  = (const float*)d_in[11];
  (void)in_sizes; (void)n_in; (void)out_size; (void)ws_size;

  char* ws = (char*)d_ws;
  size_t off = 0;
  auto take = [&](size_t bytes) -> void* {
    void* p = ws + off;
    off += (bytes + 255) & ~(size_t)255;
    return p;
  };
  __bf16* xb    = (__bf16*)take((size_t)NB * NN * ND * 2);
  __bf16* wqkvb = (__bf16*)take((size_t)NQKV * ND * 2);
  __bf16* rbw2b = (__bf16*)take((size_t)NH * ND * 2);
  float*  w2t   = (float*) take((size_t)ND * ND * 4);
  __bf16* Qb    = (__bf16*)take((size_t)NB * NH * NN * NHD * 2);
  __bf16* Kb    = (__bf16*)take((size_t)NB * NH * NN * NHD * 2);
  __bf16* Vb    = (__bf16*)take((size_t)NB * NH * NN * NHD * 2);
  float*  Sg    = (float*) take((size_t)NB * NH * NN * NN * 4);
  __bf16* Wgt   = (__bf16*)take((size_t)NB * NH * NN * NN * 2);
  float*  baseO = (float*) take((size_t)NB * NH * NN * NHD * 4);

  pa_convert<<<768, 256, 0, stream>>>(x, qkv_w, rb_w2, rv_w2, xb, wqkvb, rbw2b, w2t);
  pa_qkv<<<NB * 24 * 48, 32, 0, stream>>>(xb, wqkvb, qkv_b, Qb, Kb, Vb);
  pa_scores<<<NB * 24 * 24, 256, 0, stream>>>(coords, rb_w1, rb_b1, rb_b2, rbw2b,
                                              Qb, Kb, Sg);
  pa_softmax<<<(NB * NH * NN) / 8, 256, 0, stream>>>(Sg, Wgt);
  pa_base<<<NB * NH * 24, 64, 0, stream>>>(Wgt, Vb, baseO);
  pa_relctx<<<NB * NN, 256, 0, stream>>>(coords, rv_w1, rv_b1, rv_b2, w2t, Wgt,
                                         baseO, (float*)d_out);
}
